// BahdanauAttention_28905129902200
// MI455X (gfx1250) — matrix operand order repaired and checked by reading, not compiled, not run on hardware
//
#include <hip/hip_runtime.h>
#include <hip/hip_bf16.h>
#include <math.h>

#define B_ 64
#define S_ 2048
#define H_ 1024

#define TS 64            // s-rows per block -> W L2 traffic = (B*S/TS)*2MB = 4 GB
#define KC 128           // K chunk staged in LDS
#define XS_STR (KC + 8)  // 136 ushorts -> 16B-slot stride 17 (odd) => conflict-free

typedef unsigned short u16;
typedef unsigned int   u32;
typedef __attribute__((ext_vector_type(16))) __bf16 v16bf;
typedef __attribute__((ext_vector_type(8)))  float  v8f;
typedef __attribute__((ext_vector_type(4)))  float  f4;
typedef __attribute__((ext_vector_type(4)))  u32    u32x4;
typedef __attribute__((ext_vector_type(2)))  u32    u32x2;

__device__ __forceinline__ u16 f2bf(float x) {
  union { float f; u32 u; } c; c.f = x;
  u32 r = c.u + 0x7fffu + ((c.u >> 16) & 1u);   // round-to-nearest-even
  return (u16)(r >> 16);
}

// CDNA5 hardware tanh (transcendental VALU op)
__device__ __forceinline__ float fast_tanh(float x) {
  float r;
  asm("v_tanh_f32 %0, %1" : "=v"(r) : "v"(x));
  return r;
}

// ---- W_enc fp32 -> bf16 (2 MB, stays L2-resident for the whole GEMM) ----
__global__ __launch_bounds__(256)
void convert_w_kernel(const float* __restrict__ W, u16* __restrict__ Wb) {
  int i = (blockIdx.x * 256 + threadIdx.x) * 4;
  f4 f = *(const f4*)(W + i);
  u32x2 p = { (u32)f2bf(f.x) | ((u32)f2bf(f.y) << 16),
              (u32)f2bf(f.z) | ((u32)f2bf(f.w) << 16) };
  *(u32x2*)(Wb + i) = p;
}

// ---- dec_proj[b][o] = dec_hidden[b] . W_dec[o] ----
__global__ __launch_bounds__(256)
void dec_proj_kernel(const float* __restrict__ dec, const float* __restrict__ Wd,
                     float* __restrict__ dproj) {
  int idx = blockIdx.x * 256 + threadIdx.x;       // 65536 = B*H
  int b = idx >> 10, o = idx & (H_ - 1);
  const float* dr = dec + b * H_;
  const float* wr = Wd + (size_t)o * H_;
  float s = 0.f;
  for (int h = 0; h < H_; h += 4) {
    f4 a = *(const f4*)(dr + h);
    f4 w = *(const f4*)(wr + h);
    s += a.x * w.x + a.y * w.y + a.z * w.z + a.w * w.w;
  }
  dproj[idx] = s;
}

// ---- Fused scores: v . tanh(X @ Wenc^T + dproj) via bf16 WMMA ----
// Block: 256 threads = 8 waves, each wave owns a disjoint o-chunk of 16
// (x8 op-passes = all 1024 o) and computes 4 s-subtiles against each B
// fragment (4x B register reuse). A fragments register-resident per K-chunk.
__global__ __launch_bounds__(256)
void scores_kernel(const float* __restrict__ enc, const u16* __restrict__ Wb,
                   const float* __restrict__ dproj, const float* __restrict__ vvec,
                   float* __restrict__ scores) {
  __shared__ u16 Xs[TS * XS_STR];   // 64 x 136 bf16 = 17408 B
  __shared__ float red[8][TS];      // 2 KB
  int b = blockIdx.x, st = blockIdx.y;    // st 0..31
  int s_base = st * TS;
  int tid = threadIdx.x;
  int lane = tid & 31, wv = tid >> 5;     // wv = o-wave 0..7
  int n = lane & 15, hf = lane >> 4;

  v8f acc[8][4];                           // [op][s-subtile] = 256 VGPRs
  {
    v8f z = {0.f, 0.f, 0.f, 0.f, 0.f, 0.f, 0.f, 0.f};
    #pragma unroll
    for (int op = 0; op < 8; ++op)
      #pragma unroll
      for (int t = 0; t < 4; ++t) acc[op][t] = z;
  }

  const float* xbase = enc + ((size_t)b * S_ + s_base) * H_;
  // B fragment: lane holds column N=n -> row o of Wenc; K +8hf and +16+8hf by half
  const u16* wbase = Wb + (size_t)(wv * 16 + n) * H_ + (hf << 3);

  for (int kc = 0; kc < H_; kc += KC) {
    __syncthreads();   // protect Xs from previous chunk's readers
    // Stage X: 64 rows x 128 K fp32 -> bf16 (2048 f4 chunks, 8/thread, coalesced)
    #pragma unroll
    for (int i = tid; i < TS * (KC / 4); i += 256) {
      int r = i >> 5;                 // 32 f4-chunks per row
      int c = (i & 31) << 2;
      f4 f = __builtin_nontemporal_load((const f4*)(xbase + (size_t)r * H_ + kc + c));
      u32x2 p = { (u32)f2bf(f.x) | ((u32)f2bf(f.y) << 16),
                  (u32)f2bf(f.z) | ((u32)f2bf(f.w) << 16) };
      *(u32x2*)&Xs[r * XS_STR + c] = p;
    }
    __syncthreads();

    // A fragments: 4 s-subtiles x 4 k-steps into registers (128 VGPRs),
    // shared by all 8 op-passes below (no barriers in between).
    union Frag { u32x4 q[2]; v16bf bv; };
    Frag Af[4][4];
    #pragma unroll
    for (int t = 0; t < 4; ++t) {
      const u16* xr = Xs + (t * 16 + n) * XS_STR + (hf << 3);
      #pragma unroll
      for (int ks = 0; ks < 4; ++ks) {
        Af[t][ks].q[0] = *(const u32x4*)(xr + ks * 32);        // K +8hf
        Af[t][ks].q[1] = *(const u32x4*)(xr + ks * 32 + 16);   // K +16+8hf
      }
    }

    #pragma unroll
    for (int op = 0; op < 8; ++op) {          // o = op*128 + wv*16 + n
      const u16* wr = wbase + (size_t)op * 128 * H_ + kc;
      #pragma unroll
      for (int ks = 0; ks < 4; ++ks) {
        Frag Bf;
        Bf.q[0] = *(const u32x4*)(wr + ks * 32);       // K ks*32+8hf
        Bf.q[1] = *(const u32x4*)(wr + ks * 32 + 16);
        #pragma unroll
        for (int t = 0; t < 4; ++t) {                  // 4x B reuse, independent accs
          acc[op][t] = __builtin_amdgcn_wmma_f32_16x16x32_bf16(
              false, Af[t][ks].bv, false, Bf.bv, (short)0, acc[op][t], false, false);
        }
      }
    }
  }

  // Epilogue: e = tanh(acc + dproj) * v, reduce over o (lanes + ops + waves)
  float rsum[4][8];
  #pragma unroll
  for (int t = 0; t < 4; ++t)
    #pragma unroll
    for (int j = 0; j < 8; ++j) rsum[t][j] = 0.f;
  const float* dpr = dproj + b * H_;
  #pragma unroll
  for (int op = 0; op < 8; ++op) {
    int o = op * 128 + wv * 16 + n;
    float dpv = dpr[o];
    float vv  = vvec[o];
    #pragma unroll
    for (int t = 0; t < 4; ++t) {
      #pragma unroll
      for (int j = 0; j < 8; ++j) {
        // C layout: VGPR j, lanes 0-15 -> M=j, lanes 16-31 -> M=j+8, N=lane%16
        float e = fast_tanh(acc[op][t][j] + dpv) * vv;
        e += __shfl_xor(e, 1);
        e += __shfl_xor(e, 2);
        e += __shfl_xor(e, 4);
        e += __shfl_xor(e, 8);      // sum over N within each 16-lane half
        rsum[t][j] += e;
      }
    }
  }
  if (n == 0) {
    #pragma unroll
    for (int t = 0; t < 4; ++t)
      #pragma unroll
      for (int j = 0; j < 8; ++j)
        red[wv][t * 16 + (hf << 3) + j] = rsum[t][j];
  }
  __syncthreads();
  if (tid < TS) {
    float tsum = 0.f;
    #pragma unroll
    for (int w = 0; w < 8; ++w) tsum += red[w][tid];
    scores[b * S_ + s_base + tid] = tsum;   // deterministic
  }
}

// ---- masked softmax over S, one block per batch ----
__global__ __launch_bounds__(256)
void softmax_kernel(const float* __restrict__ scores, const unsigned char* __restrict__ mask,
                    float* __restrict__ wout) {
  __shared__ float sred[8];
  __shared__ float sbc;
  int b = blockIdx.x, tid = threadIdx.x;
  const float* sr = scores + b * S_;
  const unsigned char* mr = mask + b * S_;
  float* orow = wout + b * S_;
  float m = -3.0e38f;
  for (int s = tid; s < S_; s += 256) {
    float sc = mr[s] ? sr[s] : -1.0e9f;
    m = fmaxf(m, sc);
  }
  #pragma unroll
  for (int d = 1; d < 32; d <<= 1) m = fmaxf(m, __shfl_xor(m, d));
  if ((tid & 31) == 0) sred[tid >> 5] = m;
  __syncthreads();
  if (tid == 0) {
    float mm = sred[0];
    for (int i = 1; i < 8; ++i) mm = fmaxf(mm, sred[i]);
    sbc = mm;
  }
  __syncthreads();
  float M = sbc;
  float sum = 0.f;
  for (int s = tid; s < S_; s += 256) {
    float sc = mr[s] ? sr[s] : -1.0e9f;
    float e = __expf(sc - M);
    orow[s] = e;
    sum += e;
  }
  #pragma unroll
  for (int d = 1; d < 32; d <<= 1) sum += __shfl_xor(sum, d);
  if ((tid & 31) == 0) sred[tid >> 5] = sum;
  __syncthreads();
  if (tid == 0) {
    float t = 0.f;
    for (int i = 0; i < 8; ++i) t += sred[i];
    sbc = 1.0f / t;
  }
  __syncthreads();
  float inv = sbc;
  for (int s = tid; s < S_; s += 256) orow[s] *= inv;
}

// ---- context partials: 16 S-chunks in parallel (8192 waves of streaming f4 loads) ----
__global__ __launch_bounds__(256)
void context_part_kernel(const float* __restrict__ enc, const float* __restrict__ w,
                         float* __restrict__ part) {
  int b = blockIdx.x, ck = blockIdx.y;          // ck 0..15, 128 s each
  int h = threadIdx.x << 2;
  const float* wr = w + b * S_ + ck * 128;
  const float* er = enc + ((size_t)b * S_ + ck * 128) * H_ + h;
  f4 a = {0.f, 0.f, 0.f, 0.f};
  #pragma unroll 4
  for (int s = 0; s < 128; ++s) {
    float ws = wr[s];
    f4 x = __builtin_nontemporal_load((const f4*)(er + (size_t)s * H_));
    a += x * ws;
  }
  *(f4*)(part + ((size_t)ck * B_ + b) * H_ + h) = a;
}

__global__ __launch_bounds__(256)
void context_reduce_kernel(const float* __restrict__ part, float* __restrict__ ctx) {
  int i = blockIdx.x * 256 + threadIdx.x;       // 65536 = B*H
  float t = 0.f;
  #pragma unroll
  for (int c = 0; c < 16; ++c) t += part[(size_t)c * (B_ * H_) + i];
  ctx[i] = t;                                    // deterministic
}

extern "C" void kernel_launch(void* const* d_in, const int* in_sizes, int n_in,
                              void* d_out, int out_size, void* d_ws, size_t ws_size,
                              hipStream_t stream) {
  (void)in_sizes; (void)n_in; (void)out_size; (void)ws_size;
  const float* dec          = (const float*)d_in[0];         // [B,H]
  const float* enc          = (const float*)d_in[1];         // [B,S,H]
  const unsigned char* mask = (const unsigned char*)d_in[2]; // [B,S] bool
  const float* Wenc         = (const float*)d_in[3];         // [H,H]
  const float* Wdec         = (const float*)d_in[4];         // [H,H]
  const float* vvec         = (const float*)d_in[5];         // [H]

  char* ws = (char*)d_ws;
  u16*   Wb     = (u16*)ws;                                            // 2 MB
  float* dproj  = (float*)(ws + 2u * 1024u * 1024u);                   // 256 KB
  float* scores = (float*)(ws + 2u * 1024u * 1024u + 256u * 1024u);    // 512 KB
  float* part   = (float*)(ws + 2u * 1024u * 1024u + 768u * 1024u);    // 4 MB

  float* ctx   = (float*)d_out;              // context [B,H] first
  float* attnw = (float*)d_out + B_ * H_;    // attn_weights [B,S] second

  convert_w_kernel<<<(H_ * H_) / (256 * 4), 256, 0, stream>>>(Wenc, Wb);
  dec_proj_kernel<<<(B_ * H_) / 256, 256, 0, stream>>>(dec, Wdec, dproj);
  scores_kernel<<<dim3(B_, S_ / TS), 256, 0, stream>>>(enc, Wb, dproj, vvec, scores);
  softmax_kernel<<<B_, 256, 0, stream>>>(scores, mask, attnw);
  context_part_kernel<<<dim3(B_, 16), 256, 0, stream>>>(enc, attnw, part);
  context_reduce_kernel<<<(B_ * H_) / 256, 256, 0, stream>>>(part, ctx);
}
